// ASTEncoder_59906203844827
// MI455X (gfx1250) — compile-verified, matrix operand order checked by reading, not run on hardware
//
#include <hip/hip_runtime.h>
#include <math.h>
#include <stdint.h>

// ---- problem dims (compile-time) ----
#define B_     2
#define L_     2048
#define D_     512
#define H_     8
#define ANC_H_ 4
#define P_     16
#define DK_    64
#define NL_    2
#define DFF_   2048
#define MROWS  (B_ * L_)          // 4096 rows
#define EPS_   1e-5f

typedef __attribute__((ext_vector_type(16))) __bf16 v16bf;
typedef __attribute__((ext_vector_type(8)))  float  v8f;
typedef unsigned short ushort_t;

union FragBF { v16bf bf; unsigned u[8]; };

__device__ __forceinline__ ushort_t f2bf(float f) {
  unsigned u = __float_as_uint(f);
  u += 0x7FFFu + ((u >> 16) & 1u);            // round-to-nearest-even
  return (ushort_t)(u >> 16);
}

__device__ __forceinline__ float gelu_exact(float x) {
  return 0.5f * x * (1.0f + erff(x * 0.70710678118654752f));
}

// gfx1250 async global->LDS copy (16B per lane), tracked with ASYNCcnt.
__device__ __forceinline__ void async_g2l_b128(unsigned lds_byte_off, const void* gaddr) {
  unsigned long long ga = (unsigned long long)gaddr;
  asm volatile("global_load_async_to_lds_b128 %0, %1, off"
               :: "v"(lds_byte_off), "v"(ga) : "memory");
}

// =====================================================================
// Weight prep: fp32 [K,N] -> bf16 [N,K] (transpose + RNE downconvert).
// Runs once per call; makes GEMM A/B tiles layout-identical so both can
// be staged with async b128 copies (K contiguous per row).
// =====================================================================
__global__ __launch_bounds__(256) void cvt_transpose_bf16(
    const float* __restrict__ W, ushort_t* __restrict__ Wt, int K, int N)
{
  __shared__ ushort_t s[64 * 66];
  const int tid = threadIdx.x;
  const int k0 = blockIdx.y * 64, n0 = blockIdx.x * 64;
#pragma unroll
  for (int i = 0; i < 16; ++i) {
    int e = tid + (i << 8);
    int r = e >> 6, c = e & 63;                 // r: k row, c: n col (coalesced in n)
    s[c * 66 + r] = f2bf(W[(size_t)(k0 + r) * N + (n0 + c)]);
  }
  __syncthreads();
#pragma unroll
  for (int i = 0; i < 16; ++i) {
    int e = tid + (i << 8);
    int r = e >> 6, c = e & 63;                 // r: n row, c: k col (coalesced in k)
    Wt[(size_t)(n0 + r) * K + (k0 + c)] = s[r * 66 + c];
  }
}

// =====================================================================
// Pure-bf16 WMMA GEMM, double-buffered async LDS staging.
// Y[M,N] = epilogue( A[M,K] @ Bt[N,K]^T + bias, residual )
// Block = 256 threads = 8 waves; block tile 128x64 (waves 4Mx2N);
// each wave owns a 32x32 output tile (4 accumulators) so each K=32 step
// is 8 ds_load_b128 -> 4 v_wmma (2x fragment reuse vs 16x32 tiles).
// =====================================================================
#define BTM   128
#define BTN   64
#define LDT2  72                       // padded LDS row stride (bf16), 144 B
#define TILA  (BTM * LDT2 * 2)         // 18432 B per A tile buffer
#define TILB  (BTN * LDT2 * 2)         //  9216 B per B tile buffer

template <bool GELU, bool RES, bool OUTBF>
__global__ __launch_bounds__(256) void gemm_bf16_async(
    const ushort_t* __restrict__ A,   // [M,K] bf16 row-major
    const ushort_t* __restrict__ Bt,  // [N,K] bf16 row-major (pre-transposed W)
    const float* __restrict__ bias, const float* __restrict__ Rsd,
    void* __restrict__ Yv, int Msz, int Nsz, int Ksz)
{
  __shared__ __align__(16) ushort_t sA[2 * BTM * LDT2];
  __shared__ __align__(16) ushort_t sB[2 * BTN * LDT2];

  const int tid  = threadIdx.x;
  const int wave = tid >> 5, lane = tid & 31;
  const int m0 = blockIdx.y * BTM, n0 = blockIdx.x * BTN;
  const int wm  = (wave & 3) << 5;              // wave M tile: 0,32,64,96
  const int wnb = (wave >> 2) << 5;             // wave N tile: 0 or 32

  const unsigned baseA = (unsigned)(uintptr_t)sA;   // LDS byte offsets
  const unsigned baseB = (unsigned)(uintptr_t)sB;

  // per-thread async chunks: 16 B each; A: 4 chunks (128 rows), B: 2 (64 rows)
  const int cr = tid >> 3, cc = tid & 7;        // row 0..31, chunk 0..7

  v8f acc00 = {0.f,0.f,0.f,0.f,0.f,0.f,0.f,0.f};
  v8f acc01 = acc00, acc10 = acc00, acc11 = acc00;

  // fragment addressing (ISA 7.12.2 16-bit A 16x32 layout; B mirrored)
  const int mrow = wm  + (lane & 15);
  const int nrow = wnb + (lane & 15);
  const int kb   = (lane >> 4) << 3;            // 0 or 8

  const int nt = Ksz >> 6;

  auto issue = [&](int t, int b) {
    const int k0 = t << 6;
    const size_t r32 = (size_t)32 * Ksz;
    const size_t gA  = (size_t)(m0 + cr) * Ksz + k0 + cc * 8;
    const size_t gB  = (size_t)(n0 + cr) * Ksz + k0 + cc * 8;
    const unsigned loA = (unsigned)b * TILA + (unsigned)((cr * LDT2 + cc * 8) * 2);
    const unsigned loB = (unsigned)b * TILB + (unsigned)((cr * LDT2 + cc * 8) * 2);
    const unsigned stp = 32 * LDT2 * 2;
#pragma unroll
    for (int i = 0; i < 4; ++i)
      async_g2l_b128(baseA + loA + i * stp, A + gA + i * r32);
#pragma unroll
    for (int i = 0; i < 2; ++i)
      async_g2l_b128(baseB + loB + i * stp, Bt + gB + i * r32);
  };

  issue(0, 0);
  for (int t = 0; t < nt; ++t) {
    const int buf = t & 1;
    if (t + 1 < nt) {
      issue(t + 1, buf ^ 1);                    // overlap copy of t+1 with compute of t
      asm volatile("s_wait_asynccnt 0x6" ::: "memory");  // in-order: tile t landed
    } else {
      asm volatile("s_wait_asynccnt 0x0" ::: "memory");
    }
    __syncthreads();                            // tile t visible to all waves

    const ushort_t* tA = sA + buf * (BTM * LDT2);
    const ushort_t* tB = sB + buf * (BTN * LDT2);
#pragma unroll
    for (int kc = 0; kc < 64; kc += 32) {
      FragBF a0, a1, b0, b1;
#pragma unroll
      for (int r = 0; r < 4; ++r) {
        a0.u[r]     = *(const unsigned*)&tA[ mrow       * LDT2 + kc      + kb + 2 * r];
        a0.u[r + 4] = *(const unsigned*)&tA[ mrow       * LDT2 + kc + 16 + kb + 2 * r];
        a1.u[r]     = *(const unsigned*)&tA[(mrow + 16) * LDT2 + kc      + kb + 2 * r];
        a1.u[r + 4] = *(const unsigned*)&tA[(mrow + 16) * LDT2 + kc + 16 + kb + 2 * r];
        b0.u[r]     = *(const unsigned*)&tB[ nrow       * LDT2 + kc      + kb + 2 * r];
        b0.u[r + 4] = *(const unsigned*)&tB[ nrow       * LDT2 + kc + 16 + kb + 2 * r];
        b1.u[r]     = *(const unsigned*)&tB[(nrow + 16) * LDT2 + kc      + kb + 2 * r];
        b1.u[r + 4] = *(const unsigned*)&tB[(nrow + 16) * LDT2 + kc + 16 + kb + 2 * r];
      }
      acc00 = __builtin_amdgcn_wmma_f32_16x16x32_bf16(false, a0.bf, false, b0.bf,
                                                      (short)0, acc00, false, false);
      acc01 = __builtin_amdgcn_wmma_f32_16x16x32_bf16(false, a0.bf, false, b1.bf,
                                                      (short)0, acc01, false, false);
      acc10 = __builtin_amdgcn_wmma_f32_16x16x32_bf16(false, a1.bf, false, b0.bf,
                                                      (short)0, acc10, false, false);
      acc11 = __builtin_amdgcn_wmma_f32_16x16x32_bf16(false, a1.bf, false, b1.bf,
                                                      (short)0, acc11, false, false);
    }
    __syncthreads();                            // all reads of buf done before reuse
  }

  // ---- epilogue: C/D layout (lane = N col, VGPR r -> M = r + 8*(lane>=16)) ----
  const int mB0 = m0 + wm + ((lane >> 4) << 3);
  const int nA  = n0 + wnb + (lane & 15);
  const int nB  = nA + 16;
  const float bsA = bias[nA], bsB = bias[nB];
#pragma unroll
  for (int r = 0; r < 8; ++r) {
#pragma unroll
    for (int mi = 0; mi < 2; ++mi) {
      const int mm = mB0 + r + mi * 16;
      float y0 = (mi ? acc10[r] : acc00[r]) + bsA;
      float y1 = (mi ? acc11[r] : acc01[r]) + bsB;
      if (GELU) { y0 = gelu_exact(y0); y1 = gelu_exact(y1); }
      if (RES)  { y0 += Rsd[(size_t)mm * Nsz + nA]; y1 += Rsd[(size_t)mm * Nsz + nB]; }
      if (OUTBF) {
        ((ushort_t*)Yv)[(size_t)mm * Nsz + nA] = f2bf(y0);
        ((ushort_t*)Yv)[(size_t)mm * Nsz + nB] = f2bf(y1);
      } else {
        ((float*)Yv)[(size_t)mm * Nsz + nA] = y0;
        ((float*)Yv)[(size_t)mm * Nsz + nB] = y1;
      }
    }
  }
}

// =====================================================================
// LayerNorm over D=512; one wave per row. OBF=true emits bf16 (GEMM input).
// =====================================================================
template <bool OBF>
__global__ __launch_bounds__(256) void ln_kernel(
    const float* __restrict__ X, const float* __restrict__ g,
    const float* __restrict__ b, void* __restrict__ Yv, int rows)
{
  const int wave = threadIdx.x >> 5, lane = threadIdx.x & 31;
  const int row = blockIdx.x * 8 + wave;
  if (row >= rows) return;
  const float* x = X + (size_t)row * D_;
  float vals[16];
  float s = 0.f;
#pragma unroll
  for (int i = 0; i < 16; ++i) { vals[i] = x[lane + 32 * i]; s += vals[i]; }
#pragma unroll
  for (int off = 16; off; off >>= 1) s += __shfl_xor(s, off, 32);
  const float mean = s * (1.f / (float)D_);
  float vs = 0.f;
#pragma unroll
  for (int i = 0; i < 16; ++i) { float d = vals[i] - mean; vs += d * d; }
#pragma unroll
  for (int off = 16; off; off >>= 1) vs += __shfl_xor(vs, off, 32);
  const float rstd = rsqrtf(vs * (1.f / (float)D_) + EPS_);
#pragma unroll
  for (int i = 0; i < 16; ++i) {
    int c = lane + 32 * i;
    float y = (vals[i] - mean) * rstd * g[c] + b[c];
    if (OBF) ((ushort_t*)Yv)[(size_t)row * D_ + c] = f2bf(y);
    else     ((float*)Yv)[(size_t)row * D_ + c]    = y;
  }
}

// =====================================================================
// Relative-position gather attention. One wave per (b,h,l); lane = 2 of DK.
// scores(p) = (q+rq_p).(k_gather+rk_p)/8; softmax over P=16;
// out = sum_p a_p * (v_gather + rv_p). Output bf16 (feeds O-proj GEMM).
// =====================================================================
__global__ __launch_bounds__(256) void rel_attn_kernel(
    const float* __restrict__ q, const float* __restrict__ k,
    const float* __restrict__ v, const int* __restrict__ anc,
    const int* __restrict__ sib, const float* __restrict__ rq,
    const float* __restrict__ rk, const float* __restrict__ rv,
    ushort_t* __restrict__ ctx)
{
  const int gw   = (blockIdx.x * blockDim.x + threadIdx.x) >> 5;  // (b,h,l) id
  const int lane = threadIdx.x & 31;
  const int l = gw & (L_ - 1);
  const int h = (gw >> 11) & (H_ - 1);
  const int b = gw >> 14;
  const int* edges = (h < ANC_H_) ? anc : sib;

  const int d2 = lane * 2;
  const float2 q2 = *(const float2*)(q + (size_t)(b * L_ + l) * D_ + h * DK_ + d2);

  int   sn[P_];
  float sc[P_];
  float mx = -3.0e38f;
#pragma unroll
  for (int p = 0; p < P_; ++p) {
    sn[p] = edges[((b * P_ + p) * L_) + l];
    const float2 k2  = *(const float2*)(k  + (size_t)(b * L_ + sn[p]) * D_ + h * DK_ + d2);
    const float2 rk2 = *(const float2*)(rk + ((h * P_ + p) * DK_) + d2);
    const float2 rq2 = *(const float2*)(rq + ((h * P_ + p) * DK_) + d2);
    float s = (q2.x + rq2.x) * (k2.x + rk2.x) + (q2.y + rq2.y) * (k2.y + rk2.y);
#pragma unroll
    for (int off = 16; off; off >>= 1) s += __shfl_xor(s, off, 32);
    s *= 0.125f;                               // 1/sqrt(64)
    sc[p] = s;
    mx = fmaxf(mx, s);
  }
  float den = 0.f;
#pragma unroll
  for (int p = 0; p < P_; ++p) { sc[p] = expf(sc[p] - mx); den += sc[p]; }
  const float inv = 1.f / den;
  float ox = 0.f, oy = 0.f;
#pragma unroll
  for (int p = 0; p < P_; ++p) {
    const float2 v2  = *(const float2*)(v  + (size_t)(b * L_ + sn[p]) * D_ + h * DK_ + d2);
    const float2 rv2 = *(const float2*)(rv + ((h * P_ + p) * DK_) + d2);
    const float a = sc[p] * inv;
    ox += a * (v2.x + rv2.x);
    oy += a * (v2.y + rv2.y);
  }
  unsigned pk = (unsigned)f2bf(ox) | ((unsigned)f2bf(oy) << 16);
  *(unsigned*)(ctx + (size_t)(b * L_ + l) * D_ + h * DK_ + d2) = pk;
}

// =====================================================================
// Host-side orchestration
// =====================================================================
extern "C" void kernel_launch(void* const* d_in, const int* in_sizes, int n_in,
                              void* d_out, int out_size, void* d_ws, size_t ws_size,
                              hipStream_t stream) {
  const float* emb   = (const float*)d_in[0];
  const int*   anc   = (const int*)  d_in[1];
  const int*   sib   = (const int*)  d_in[2];
  const float* rel_q = (const float*)d_in[3];
  const float* rel_k = (const float*)d_in[4];
  const float* rel_v = (const float*)d_in[5];
  const float* Wq = (const float*)d_in[6];  const float* bq = (const float*)d_in[7];
  const float* Wk = (const float*)d_in[8];  const float* bk = (const float*)d_in[9];
  const float* Wv = (const float*)d_in[10]; const float* bv = (const float*)d_in[11];
  const float* Wo = (const float*)d_in[12]; const float* bo = (const float*)d_in[13];
  const float* ln1_g = (const float*)d_in[14]; const float* ln1_b = (const float*)d_in[15];
  const float* ln2_g = (const float*)d_in[16]; const float* ln2_b = (const float*)d_in[17];
  const float* W1 = (const float*)d_in[18]; const float* b1 = (const float*)d_in[19];
  const float* W2 = (const float*)d_in[20]; const float* b2 = (const float*)d_in[21];
  const float* lnf_g = (const float*)d_in[22]; const float* lnf_b = (const float*)d_in[23];

  // ---- workspace layout (56 MB total) ----
  const size_t SZ  = (size_t)MROWS * D_;         // 2,097,152 elements
  const size_t WDD = (size_t)D_ * D_;            // 262,144
  const size_t WDF = (size_t)D_ * DFF_;          // 1,048,576
  ushort_t* wqT = (ushort_t*)d_ws;               // [N=D,K=D] per layer
  ushort_t* wkT = wqT + NL_ * WDD;
  ushort_t* wvT = wkT + NL_ * WDD;
  ushort_t* woT = wvT + NL_ * WDD;
  ushort_t* w1T = woT + NL_ * WDD;               // [N=DFF,K=D] per layer
  ushort_t* w2T = w1T + NL_ * WDF;               // [N=D,K=DFF] per layer
  float*    xA  = (float*)(w2T + NL_ * WDF);     // fp32 activations
  float*    xB  = xA + SZ;
  ushort_t* xn  = (ushort_t*)(xB + SZ);          // bf16 LN output
  ushort_t* ctx = xn;                            // alias (safe by stream order)
  float*    qb  = (float*)(xn + SZ);
  float*    kb_ = qb + SZ;
  float*    vb  = kb_ + SZ;
  ushort_t* ffb = (ushort_t*)qb;                 // bf16 [M,DFF], aliases q+k region

  const dim3 blk(256);
  const dim3 gD(D_ / BTN,   MROWS / BTM);        // (8, 32)
  const dim3 gF(DFF_ / BTN, MROWS / BTM);        // (32, 32)
  const int  lnGrid = MROWS / 8;                 // 512
  const int  atGrid = (B_ * H_ * L_) / 8;        // 4096

  // ---- weight prep: downconvert + transpose once per call ----
  for (int i = 0; i < NL_; ++i) {
    cvt_transpose_bf16<<<dim3(D_/64,   D_/64),   blk, 0, stream>>>(Wq + i*WDD, wqT + i*WDD, D_, D_);
    cvt_transpose_bf16<<<dim3(D_/64,   D_/64),   blk, 0, stream>>>(Wk + i*WDD, wkT + i*WDD, D_, D_);
    cvt_transpose_bf16<<<dim3(D_/64,   D_/64),   blk, 0, stream>>>(Wv + i*WDD, wvT + i*WDD, D_, D_);
    cvt_transpose_bf16<<<dim3(D_/64,   D_/64),   blk, 0, stream>>>(Wo + i*WDD, woT + i*WDD, D_, D_);
    cvt_transpose_bf16<<<dim3(DFF_/64, D_/64),   blk, 0, stream>>>(W1 + i*WDF, w1T + i*WDF, D_, DFF_);
    cvt_transpose_bf16<<<dim3(D_/64,   DFF_/64), blk, 0, stream>>>(W2 + i*WDF, w2T + i*WDF, DFF_, D_);
  }

  // ---- encoder layers ----
  const float* x_in = emb;
  for (int i = 0; i < NL_; ++i) {
    // sublayer 1: pre-norm relative self-attention
    ln_kernel<true><<<lnGrid, blk, 0, stream>>>(x_in, ln1_g + i*D_, ln1_b + i*D_, xn, MROWS);
    gemm_bf16_async<false, false, false><<<gD, blk, 0, stream>>>(
        xn, wqT + i*WDD, bq + i*D_, nullptr, qb, MROWS, D_, D_);
    gemm_bf16_async<false, false, false><<<gD, blk, 0, stream>>>(
        xn, wkT + i*WDD, bk + i*D_, nullptr, kb_, MROWS, D_, D_);
    gemm_bf16_async<false, false, false><<<gD, blk, 0, stream>>>(
        xn, wvT + i*WDD, bv + i*D_, nullptr, vb, MROWS, D_, D_);
    rel_attn_kernel<<<atGrid, blk, 0, stream>>>(qb, kb_, vb, anc, sib,
                                                rel_q, rel_k, rel_v, ctx);
    gemm_bf16_async<false, true, false><<<gD, blk, 0, stream>>>(
        ctx, woT + i*WDD, bo + i*D_, x_in, xA, MROWS, D_, D_);
    // sublayer 2: pre-norm feed-forward (exact gelu)
    ln_kernel<true><<<lnGrid, blk, 0, stream>>>(xA, ln2_g + i*D_, ln2_b + i*D_, xn, MROWS);
    gemm_bf16_async<true, false, true><<<gF, blk, 0, stream>>>(
        xn, w1T + i*WDF, b1 + i*DFF_, nullptr, ffb, MROWS, DFF_, D_);
    gemm_bf16_async<false, true, false><<<gD, blk, 0, stream>>>(
        ffb, w2T + i*WDF, b2 + i*D_, xA, xB, MROWS, D_, DFF_);
    x_in = xB;
  }
  // final norm -> d_out (fp32)
  ln_kernel<false><<<lnGrid, blk, 0, stream>>>(x_in, lnf_g, lnf_b, d_out, MROWS);
}
